// KDAPolicyNetwork_27058293965532
// MI455X (gfx1250) — matrix operand order repaired and checked by reading, not compiled, not run on hardware
//
#include <hip/hip_runtime.h>
#include <math.h>

typedef __attribute__((ext_vector_type(16))) _Float16 v16h;
typedef __attribute__((ext_vector_type(8)))  _Float16 v8h;
typedef __attribute__((ext_vector_type(8)))  float    v8f;
typedef _Float16 h16;

#define BB 8
#define NM 4
#define TT 1024
#define DD 256
#define ND 1024
#define BT 8192
#define EE 16
#define DK 16
#define DV 16
#define DKP 32
#define RR 4
#define DA 25
#define EPSF 1.1920929e-07f

__device__ __forceinline__ float sigf(float x){ return 1.f/(1.f+__expf(-x)); }
__device__ __forceinline__ float siluf(float x){ return x*sigf(x); }

// CDNA5 async global->LDS (ASYNCcnt-tracked), 16B per lane.
__device__ __forceinline__ void async_ld_b128(unsigned lds_off, const void* gaddr) {
  asm volatile("global_load_async_to_lds_b128 %0, %1, off" :: "v"(lds_off), "v"(gaddr) : "memory");
}
__device__ __forceinline__ void async_wait0() {
  asm volatile("s_wait_asynccnt 0x0" ::: "memory");
}

// ---------------- generic WMMA GEMM: C(M,N) = A(M,K) * B(K,N) ----------------
// A: f16 row-major (lda). Bt: f16 K-major per output column (row n = B[:,n], ldbt >= K).
// One wave computes a 16 x (16*NT) strip: A fragment reused NT times from registers.
// Grid: (N/(16*NT), M/16, batch), block = 32 (one wave).
template <int NT>
__global__ void gemm_wmma_f16(const h16* __restrict__ A, const h16* __restrict__ Bt,
                              float* __restrict__ C, int K, int lda, int ldbt, int ldc,
                              long sA, long sB, long sC) {
  const h16* Ab = A + (long)blockIdx.z * sA;
  const h16* Bb = Bt + (long)blockIdx.z * sB;
  float*     Cb = C + (long)blockIdx.z * sC;
  int ng = blockIdx.x, mt = blockIdx.y;
  int lane = threadIdx.x;
  int hf = lane >> 4;       // lane half (0/1)
  int mn = lane & 15;       // row for A, col for B/C
  v8f acc[NT];
#pragma unroll
  for (int nt = 0; nt < NT; ++nt) acc[nt] = (v8f){};
  const h16* arow  = Ab + (long)(mt * 16 + mn) * lda;
  const h16* brow0 = Bb + (long)(ng * NT * 16 + mn) * ldbt;
  for (int kk = 0; kk < K; kk += 32) {
    __builtin_prefetch(arow + kk + 32, 0, 1);          // -> global_prefetch_b8
    v8h a0 = *(const v8h*)(arow + kk + hf * 16);       // K = 16h .. 16h+7
    v8h a1 = *(const v8h*)(arow + kk + hf * 16 + 8);   // K = 16h+8 .. 16h+15
    v16h av;
#pragma unroll
    for (int i = 0; i < 8; ++i) { av[i] = a0[i]; av[8 + i] = a1[i]; }
#pragma unroll
    for (int nt = 0; nt < NT; ++nt) {
      v16h bv = *(const v16h*)(brow0 + (long)nt * 16 * ldbt + kk + hf * 16); // K = 16h..16h+15
      acc[nt] = __builtin_amdgcn_wmma_f32_16x16x32_f16(false, av, false, bv,
                                                       (short)0, acc[nt], false, false);
    }
  }
#pragma unroll
  for (int nt = 0; nt < NT; ++nt) {
#pragma unroll
    for (int r = 0; r < 8; ++r) {
      int m = mt * 16 + r + 8 * hf;
      Cb[(long)m * ldc + (ng * NT + nt) * 16 + mn] = acc[nt][r];
    }
  }
}

// ---------------- weight prep ----------------
__global__ void fold_mhc_kernel(const float* __restrict__ norm_w, const float* __restrict__ ppre,
                                const float* __restrict__ ppost, const float* __restrict__ pres,
                                h16* __restrict__ Bt) {
  int idx = blockIdx.x * 256 + threadIdx.x;           // E*ND
  if (idx >= EE * ND) return;
  int e = idx >> 10, k = idx & 1023;
  float w = norm_w[e * ND + k];
  for (int n = 0; n < 4; ++n) {
    Bt[(long)(e * 24 + n)     * ND + k] = (h16)(w * ppre [((long)e * ND + k) * 4 + n]);
    Bt[(long)(e * 24 + 4 + n) * ND + k] = (h16)(w * ppost[((long)e * ND + k) * 4 + n]);
  }
  for (int m = 0; m < 16; ++m)
    Bt[(long)(e * 24 + 8 + m) * ND + k] = (h16)(w * pres[((long)e * ND + k) * 16 + m]);
}

__global__ void wcat_q_kernel(const float* __restrict__ Wq, const float* __restrict__ laq,
                              h16* __restrict__ Bt) {
  int idx = blockIdx.x * 256 + threadIdx.x;           // E*32*256
  if (idx >= EE * 32 * DD) return;
  int d = idx & 255, c = (idx >> 8) & 31, e = idx >> 13;
  float v = 0.f;
  if (c < 16)      v = Wq[d * DK + c];
  else if (c < 20) v = laq[(long)e * DD * RR + d * RR + (c - 16)];
  Bt[(long)e * 32 * DD + (long)c * DD + d] = (h16)v;
}

__global__ void wcat_kv_kernel(const float* __restrict__ Wk, const float* __restrict__ Wv,
                               const float* __restrict__ lak, const float* __restrict__ lav,
                               const float* __restrict__ aup, const float* __restrict__ bup,
                               h16* __restrict__ Bt) {
  int idx = blockIdx.x * 256 + threadIdx.x;           // E*96*256
  if (idx >= EE * 96 * DD) return;
  int d = idx & 255, c = (idx >> 8) % 96, e = idx / (96 * 256);
  float v = 0.f;
  if (c < 16)      v = Wk[d * DK + c];
  else if (c < 32) v = Wv[d * DV + (c - 16)];
  else if (c < 36) v = lak[(long)e * DD * RR + d * RR + (c - 32)];
  else if (c < 40) v = lav[(long)e * DD * RR + d * RR + (c - 36)];
  else if (c < 65) v = aup[(long)e * DD * DA + d * DA + (c - 40)];
  else if (c < 90) v = bup[(long)e * DD * DA + d * DA + (c - 65)];
  Bt[(long)e * 96 * DD + (long)c * DD + d] = (h16)v;
}

__global__ void wtail_kernel(const float* __restrict__ W_pre, const float* __restrict__ W_o,
                             const float* __restrict__ W_pg1, const float* __restrict__ W_pg2,
                             h16* __restrict__ Bpre, h16* __restrict__ Bo,
                             h16* __restrict__ Bpg1, h16* __restrict__ Bpg2) {
  int idx = blockIdx.x * 256 + threadIdx.x;
  if (idx < 65536) {
    int n = idx >> 8, kd = idx & 255;
    Bpre[(long)n * 256 + kd] = (h16)W_pre[kd * 256 + n];
    Bo  [(long)n * 256 + kd] = (h16)W_o  [kd * 256 + n];
    if (kd < 160) Bpg2[(long)n * 160 + kd] = (h16)((kd < 158) ? W_pg2[kd * 256 + n] : 0.f);
  }
  if (idx < 40960) {                                   // Bpg1: N=160 (pad 158->160), K=256
    int n = idx / 256, d = idx % 256;
    Bpg1[(long)n * 256 + d] = (h16)((n < 158) ? W_pg1[d * 158 + n] : 0.f);
  }
}

// ---------------- RMS + router gates (stream staged via async-to-LDS) ----------------
__global__ void rms_route_kernel(const float* __restrict__ st, const float* __restrict__ rqw,
                                 const float* __restrict__ rkw, h16* __restrict__ rx,
                                 float* __restrict__ gate_q, float* __restrict__ gate_kv) {
  __shared__ float xs[ND];
  __shared__ float red[256];
  __shared__ float lg[32];
  long bt = blockIdx.x; int b = bt >> 10, t = bt & 1023, tid = threadIdx.x;
  {
    int n0 = tid >> 6, o0 = (tid & 63) * 4;            // k = tid*4 -> (n, d)
    const float* g = &st[(((long)(b * 4 + n0) * TT) + t) * DD + o0];
    unsigned lds0 = (unsigned)(size_t)(&xs[0]);        // LDS aperture: offset = addr[31:0]
    async_ld_b128(lds0 + (unsigned)tid * 16u, g);
    async_wait0();
  }
  __syncthreads();
  float loc = 0.f;
  for (int n = 0; n < 4; ++n) { float v = xs[n * 256 + tid]; loc += v * v; }
  red[tid] = loc; __syncthreads();
  for (int s = 128; s > 0; s >>= 1) { if (tid < s) red[tid] += red[tid + s]; __syncthreads(); }
  float scale = rsqrtf(red[0] * (1.f / (float)ND) + EPSF);
  for (int n = 0; n < 4; ++n)
    rx[bt * ND + n * 256 + tid] = (h16)(xs[n * 256 + tid] * scale);
  if (tid < 32) {
    int e = tid & 15;
    const float* w = (tid < 16) ? rqw : rkw;
    float acc = 0.f;
    for (int d = 0; d < 256; ++d) {
      float m = 0.25f * (xs[d] + xs[256 + d] + xs[512 + d] + xs[768 + d]);
      acc += m * w[d * 16 + e];
    }
    lg[tid] = acc;
  }
  __syncthreads();
  if (tid < 2) {
    const float* L = &lg[tid * 16];
    float* out = ((tid == 0) ? gate_q : gate_kv) + bt * 16;
    float mx = L[0]; for (int i = 1; i < 16; ++i) mx = fmaxf(mx, L[i]);
    float p[16], sum = 0.f;
    for (int i = 0; i < 16; ++i) { p[i] = __expf(L[i] - mx); sum += p[i]; }
    for (int i = 0; i < 16; ++i) p[i] /= sum;
    int ord[16]; for (int i = 0; i < 16; ++i) ord[i] = i;
    for (int i = 0; i < 16; ++i)
      for (int j = i + 1; j < 16; ++j)
        if (p[ord[j]] > p[ord[i]]) { int tm = ord[i]; ord[i] = ord[j]; ord[j] = tm; }
    float cs = 0.f, sel[16];
    for (int i = 0; i < 16; ++i) sel[i] = 0.f;
    for (int i = 0; i < 16; ++i) {
      float sp = p[ord[i]];
      bool m = ((cs < 0.8f) && (i < 4)) || (i == 0);
      cs += sp;
      if (m) sel[ord[i]] = sp;
    }
    for (int i = 0; i < 16; ++i) out[i] = sel[i];
  }
}

// ---------------- mHC nonlinearity + sinkhorn (in place on GEMM output) ----------------
__global__ void mhc_post_kernel(float* __restrict__ mhc, const float* __restrict__ b_pre,
                                const float* __restrict__ b_post, const float* __restrict__ b_res,
                                const float* __restrict__ a_pre, const float* __restrict__ a_post,
                                const float* __restrict__ a_res) {
  long idx = (long)blockIdx.x * 256 + threadIdx.x;    // BT*E
  if (idx >= (long)BT * EE) return;
  int e = (int)(idx & 15); long bt = idx >> 4;
  float* p = mhc + bt * 384 + e * 24;
  float ap = a_pre[e], apo = a_post[e], ar = a_res[e];
  for (int n = 0; n < 4; ++n) p[n]     = sigf(ap * p[n] + b_pre[e * 4 + n]);
  for (int n = 0; n < 4; ++n) p[4 + n] = 2.f * sigf(apo * p[4 + n] + b_post[e * 4 + n]);
  float M[16];
  for (int i = 0; i < 16; ++i) M[i] = __expf(ar * p[8 + i] + b_res[e * 16 + i]);
  for (int it = 0; it < 6; ++it) {
    for (int i = 0; i < 4; ++i) {
      float s = M[i*4] + M[i*4+1] + M[i*4+2] + M[i*4+3];
      for (int j = 0; j < 4; ++j) M[i*4+j] /= s;
    }
    for (int j = 0; j < 4; ++j) {
      float s = M[j] + M[4+j] + M[8+j] + M[12+j];
      for (int i = 0; i < 4; ++i) M[i*4+j] /= s;
    }
  }
  for (int i = 0; i < 16; ++i) p[8 + i] = M[i];
}

// ---------------- h = Hpre @ stream (+ routed combos), stream via async-to-LDS ----------------
__global__ void compute_h_kernel(const float* __restrict__ st, const float* __restrict__ mhcq,
                                 const float* __restrict__ mhckv, const float* __restrict__ gate_q,
                                 const float* __restrict__ gate_kv, h16* __restrict__ h_q,
                                 h16* __restrict__ h_kv, h16* __restrict__ hq_r,
                                 h16* __restrict__ hkv_r, float* __restrict__ Hpost_r,
                                 float* __restrict__ Hres_r) {
  __shared__ float xs[ND];
  __shared__ float preQ[64], preKV[64], postKV[64], gq[16], gk[16];
  long bt = blockIdx.x; int b = bt >> 10, t = bt & 1023, tid = threadIdx.x;
  {
    int n0 = tid >> 6, o0 = (tid & 63) * 4;
    const float* g = &st[(((long)(b * 4 + n0) * TT) + t) * DD + o0];
    unsigned lds0 = (unsigned)(size_t)(&xs[0]);
    async_ld_b128(lds0 + (unsigned)tid * 16u, g);
  }
  if (tid < 64) {
    int e = tid >> 2, n = tid & 3;
    preQ[tid]   = mhcq [bt * 384 + e * 24 + n];
    preKV[tid]  = mhckv[bt * 384 + e * 24 + n];
    postKV[tid] = mhckv[bt * 384 + e * 24 + 4 + n];
  }
  if (tid >= 64 && tid < 80) gq[tid - 64] = gate_q[bt * 16 + tid - 64];
  if (tid >= 80 && tid < 96) gk[tid - 80] = gate_kv[bt * 16 + tid - 80];
  async_wait0();
  __syncthreads();
  int d = tid;
  float x0 = xs[d], x1 = xs[256 + d], x2 = xs[512 + d], x3 = xs[768 + d];
  float aq = 0.f, ak = 0.f;
  for (int e = 0; e < 16; ++e) {
    float hv = preQ[e*4]*x0 + preQ[e*4+1]*x1 + preQ[e*4+2]*x2 + preQ[e*4+3]*x3;
    h_q[((long)e * BT + bt) * DD + d] = (h16)hv;  aq += gq[e] * hv;
    float hk = preKV[e*4]*x0 + preKV[e*4+1]*x1 + preKV[e*4+2]*x2 + preKV[e*4+3]*x3;
    h_kv[((long)e * BT + bt) * DD + d] = (h16)hk; ak += gk[e] * hk;
  }
  hq_r[bt * DD + d] = (h16)aq;
  hkv_r[bt * DD + d] = (h16)ak;
  if (tid < 4) {
    float s = 0.f; for (int e = 0; e < 16; ++e) s += gk[e] * postKV[e * 4 + tid];
    Hpost_r[bt * 4 + tid] = s;
  }
  if (tid >= 4 && tid < 20) {
    int ij = tid - 4; float s = 0.f;
    for (int e = 0; e < 16; ++e) s += gk[e] * mhckv[bt * 384 + e * 24 + 8 + ij];
    Hres_r[bt * 16 + ij] = s;
  }
}

// ---------------- per-(b,t) expert epilogues ----------------
__global__ void epi_q_kernel(const float* __restrict__ proj_q, const float* __restrict__ gate_q,
                             const float* __restrict__ lora_B_q, float* __restrict__ q_total) {
  long bt = (long)blockIdx.x * 64 + threadIdx.x;
  if (bt >= BT) return;
  float t = (float)(bt & 1023);
  float fr[16];
  for (int k = 0; k < 16; ++k) fr[k] = __powf(10000.f, (float)k * (1.f / 16.f));
  float qt[32]; for (int i = 0; i < 32; ++i) qt[i] = 0.f;
  for (int e = 0; e < 16; ++e) {
    float g = gate_q[bt * 16 + e];
    if (g == 0.f) continue;
    const float* pr = proj_q + ((long)e * BT + bt) * 32;
    float lin[4];
    for (int r = 0; r < 4; ++r) lin[r] = siluf(pr[16 + r]);
    float raw[16], ss = 0.f;
    for (int k = 0; k < 16; ++k) {
      float v = pr[k];
      for (int r = 0; r < 4; ++r) v += lin[r] * lora_B_q[e * 64 + r * 16 + k];
      raw[k] = v; ss += v * v;
    }
    float inv = 1.f / fmaxf(sqrtf(ss), 1e-12f);
    for (int k = 0; k < 16; ++k) {
      float mu = log1pf(__expf(raw[k] * inv));
      float ph = t * fr[k];
      qt[k]      += g * mu * cosf(ph);
      qt[16 + k] += g * mu * sinf(ph);
    }
  }
  for (int i = 0; i < 32; ++i) q_total[bt * 32 + i] = qt[i];
}

__global__ void epi_kv_kernel(const float* __restrict__ proj_kv, const float* __restrict__ gate_kv,
                              const float* __restrict__ lora_B_k, const float* __restrict__ lora_B_v,
                              const float* __restrict__ alpha_down, const float* __restrict__ beta_down,
                              const float* __restrict__ pope_delta, float* __restrict__ k_total,
                              float* __restrict__ v_total, float* __restrict__ alpha_total,
                              float* __restrict__ beta_total) {
  long bt = (long)blockIdx.x * 64 + threadIdx.x;
  if (bt >= BT) return;
  float t = (float)(bt & 1023);
  float fr[16], dph[16];
  for (int k = 0; k < 16; ++k) {
    fr[k] = __powf(10000.f, (float)k * (1.f / 16.f));
    dph[k] = 6.2831853071795864f * sigf(pope_delta[k]);
  }
  float kt[32], vt[16], at[32], btot = 0.f;
  for (int i = 0; i < 32; ++i) { kt[i] = 0.f; at[i] = 0.f; }
  for (int i = 0; i < 16; ++i) vt[i] = 0.f;
  for (int e = 0; e < 16; ++e) {
    float g = gate_kv[bt * 16 + e];
    if (g == 0.f) continue;
    const float* pr = proj_kv + ((long)e * BT + bt) * 96;
    float lk[4], lv[4];
    for (int r = 0; r < 4; ++r) { lk[r] = siluf(pr[32 + r]); lv[r] = siluf(pr[36 + r]); }
    float raw[16], ss = 0.f;
    for (int k = 0; k < 16; ++k) {
      float v = pr[k];
      for (int r = 0; r < 4; ++r) v += lk[r] * lora_B_k[e * 64 + r * 16 + k];
      raw[k] = v; ss += v * v;
    }
    float inv = 1.f / fmaxf(sqrtf(ss), 1e-12f);
    for (int k = 0; k < 16; ++k) {
      float mu = log1pf(__expf(raw[k] * inv));
      float ph = t * fr[k] - dph[k];
      kt[k]      += g * mu * cosf(ph);
      kt[16 + k] += g * mu * sinf(ph);
      float vv = pr[16 + k];
      for (int r = 0; r < 4; ++r) vv += lv[r] * lora_B_v[e * 64 + r * 16 + k];
      vt[k] += g * siluf(vv);
    }
    float sa[25], sb[25];
    for (int h = 0; h < 25; ++h) { sa[h] = siluf(pr[40 + h]); sb[h] = siluf(pr[65 + h]); }
    for (int k = 0; k < 32; ++k) {
      float s = 0.f;
      for (int h = 0; h < 25; ++h) s += sa[h] * alpha_down[e * 800 + h * 32 + k];
      at[k] += g * sigf(s);
    }
    float s = 0.f;
    for (int h = 0; h < 25; ++h) s += sb[h] * beta_down[e * 25 + h];
    btot += g * sigf(s);
  }
  for (int i = 0; i < 32; ++i) { k_total[bt * 32 + i] = kt[i]; alpha_total[bt * 32 + i] = at[i]; }
  for (int i = 0; i < 16; ++i) v_total[bt * 16 + i] = vt[i];
  beta_total[bt] = btot;
}

// ---------------- KDA delta-rule scan: one wave per batch ----------------
__global__ void kda_kernel(const float* __restrict__ q_total, const float* __restrict__ k_total,
                           const float* __restrict__ v_total, const float* __restrict__ alpha_total,
                           const float* __restrict__ beta_total, float* __restrict__ out_kda,
                           float* __restrict__ S_out) {
  __shared__ float sq[32], sk[32], sa[32], svb[17];
  int b = blockIdx.x, lane = threadIdx.x;
  float S[32];
  for (int d = 0; d < 32; ++d) S[d] = 0.f;
  for (int t = 0; t < TT; ++t) {
    long bt = (long)b * TT + t;
    sq[lane] = q_total[bt * 32 + lane];
    sk[lane] = k_total[bt * 32 + lane];
    sa[lane] = alpha_total[bt * 32 + lane];
    if (lane < 16) svb[lane] = v_total[bt * 16 + lane];
    if (lane == 16) svb[16] = beta_total[bt];
    __syncthreads();
    if (lane < 16) {
      float kaS = 0.f;
      for (int d = 0; d < 32; ++d) kaS += sk[d] * sa[d] * S[d];
      float upd = svb[16] * (svb[lane] - kaS);
      float o = 0.f;
      for (int d = 0; d < 32; ++d) { S[d] = sa[d] * S[d] + sk[d] * upd; o += sq[d] * S[d]; }
      out_kda[bt * 16 + lane] = o;
    }
    __syncthreads();
  }
  if (lane < 16)
    for (int d = 0; d < 32; ++d) S_out[((long)b * 32 + d) * 16 + lane] = S[d];
}

// ---------------- tail elementwise ----------------
__global__ void yin_kernel(const float* __restrict__ prelin, const float* __restrict__ out_kda,
                           h16* __restrict__ yin) {
  long i = (long)blockIdx.x * 256 + threadIdx.x;      // BT*256
  long bt = i >> 8; int c = (int)(i & 255);
  yin[i] = (h16)(sigf(prelin[i]) * out_kda[bt * 16 + (c & 15)]);
}

__global__ void zbuf_kernel(const float* __restrict__ pg1, h16* __restrict__ zbuf) {
  long i = (long)blockIdx.x * 256 + threadIdx.x;      // BT*160
  if (i >= (long)BT * 160) return;
  int j = (int)(i % 160);
  zbuf[i] = (h16)((j < 158) ? siluf(pg1[i]) : 0.f);
}

__global__ void ogate_kernel(const float* __restrict__ y, const float* __restrict__ pg2lin,
                             float* __restrict__ o) {
  long i = (long)blockIdx.x * 256 + threadIdx.x;      // BT*256
  o[i] = y[i] * sigf(pg2lin[i]);
}

// ---------------- stream write-back ----------------
__global__ void writeback_kernel(const float* __restrict__ st, const float* __restrict__ o,
                                 const float* __restrict__ Hres_r, const float* __restrict__ Hpost_r,
                                 float* __restrict__ out) {
  long bt = blockIdx.x; int b = bt >> 10, t = bt & 1023, d = threadIdx.x;
  float sj[4];
  for (int j = 0; j < 4; ++j) sj[j] = st[(((long)(b * 4 + j) * TT) + t) * DD + d];
  float ov = o[bt * DD + d];
  for (int i = 0; i < 4; ++i) {
    float v = Hpost_r[bt * 4 + i] * ov;
    for (int j = 0; j < 4; ++j) v += Hres_r[bt * 16 + i * 4 + j] * sj[j];
    out[(((long)(b * 4 + i) * TT) + t) * DD + d] = v;
  }
}

// ---------------- launch ----------------
extern "C" void kernel_launch(void* const* d_in, const int* in_sizes, int n_in,
                              void* d_out, int out_size, void* d_ws, size_t ws_size,
                              hipStream_t stream) {
  (void)in_sizes; (void)n_in; (void)out_size; (void)ws_size;
  const float* st        = (const float*)d_in[0];
  const float* Wq        = (const float*)d_in[1];
  const float* Wk        = (const float*)d_in[2];
  const float* Wv        = (const float*)d_in[3];
  const float* pope      = (const float*)d_in[4];
  const float* lora_A_q  = (const float*)d_in[5];
  const float* lora_B_q  = (const float*)d_in[6];
  const float* lora_A_k  = (const float*)d_in[7];
  const float* lora_B_k  = (const float*)d_in[8];
  const float* lora_A_v  = (const float*)d_in[9];
  const float* lora_B_v  = (const float*)d_in[10];
  const float* alpha_up  = (const float*)d_in[11];
  const float* alpha_dn  = (const float*)d_in[12];
  const float* beta_up   = (const float*)d_in[13];
  const float* beta_dn   = (const float*)d_in[14];
  const float* q_norm_w  = (const float*)d_in[15];
  const float* q_ppre    = (const float*)d_in[16];
  const float* q_ppost   = (const float*)d_in[17];
  const float* q_pres    = (const float*)d_in[18];
  const float* q_bpre    = (const float*)d_in[19];
  const float* q_bpost   = (const float*)d_in[20];
  const float* q_bres    = (const float*)d_in[21];
  const float* q_apre    = (const float*)d_in[22];
  const float* q_apost   = (const float*)d_in[23];
  const float* q_ares    = (const float*)d_in[24];
  const float* kv_norm_w = (const float*)d_in[25];
  const float* kv_ppre   = (const float*)d_in[26];
  const float* kv_ppost  = (const float*)d_in[27];
  const float* kv_pres   = (const float*)d_in[28];
  const float* kv_bpre   = (const float*)d_in[29];
  const float* kv_bpost  = (const float*)d_in[30];
  const float* kv_bres   = (const float*)d_in[31];
  const float* kv_apre   = (const float*)d_in[32];
  const float* kv_apost  = (const float*)d_in[33];
  const float* kv_ares   = (const float*)d_in[34];
  const float* W_pre     = (const float*)d_in[35];
  const float* W_o       = (const float*)d_in[36];
  const float* W_pg1     = (const float*)d_in[37];
  const float* W_pg2     = (const float*)d_in[38];
  const float* rqw       = (const float*)d_in[39];
  const float* rkw       = (const float*)d_in[40];

  char* wsb = (char*)d_ws;
  size_t off = 0;
  auto alloc = [&](size_t bytes) -> void* {
    void* p = wsb + off;
    off = (off + bytes + 255) & ~(size_t)255;
    return p;
  };
  h16*   rx      = (h16*)  alloc((size_t)BT * ND * 2);
  float* gate_q  = (float*)alloc((size_t)BT * 16 * 4);
  float* gate_kv = (float*)alloc((size_t)BT * 16 * 4);
  h16*   WfoldQ  = (h16*)  alloc((size_t)384 * ND * 2);
  h16*   WfoldK  = (h16*)  alloc((size_t)384 * ND * 2);
  float* mhcQ    = (float*)alloc((size_t)BT * 384 * 4);
  float* mhcK    = (float*)alloc((size_t)BT * 384 * 4);
  h16*   h_q     = (h16*)  alloc((size_t)EE * BT * DD * 2);
  h16*   h_kv    = (h16*)  alloc((size_t)EE * BT * DD * 2);
  h16*   hq_r    = (h16*)  alloc((size_t)BT * DD * 2);
  h16*   hkv_r   = (h16*)  alloc((size_t)BT * DD * 2);
  float* Hpost_r = (float*)alloc((size_t)BT * 4 * 4);
  float* Hres_r  = (float*)alloc((size_t)BT * 16 * 4);
  h16*   WcatQ   = (h16*)  alloc((size_t)EE * 32 * DD * 2);
  h16*   WcatK   = (h16*)  alloc((size_t)EE * 96 * DD * 2);
  float* proj_q  = (float*)alloc((size_t)EE * BT * 32 * 4);
  float* proj_kv = (float*)alloc((size_t)EE * BT * 96 * 4);
  float* q_tot   = (float*)alloc((size_t)BT * 32 * 4);
  float* k_tot   = (float*)alloc((size_t)BT * 32 * 4);
  float* v_tot   = (float*)alloc((size_t)BT * 16 * 4);
  float* a_tot   = (float*)alloc((size_t)BT * 32 * 4);
  float* b_tot   = (float*)alloc((size_t)BT * 4);
  float* out_kda = (float*)alloc((size_t)BT * 16 * 4);
  h16*   Bpre    = (h16*)  alloc((size_t)256 * 256 * 2);
  h16*   Bo      = (h16*)  alloc((size_t)256 * 256 * 2);
  h16*   Bpg1    = (h16*)  alloc((size_t)160 * 256 * 2);
  h16*   Bpg2    = (h16*)  alloc((size_t)256 * 160 * 2);
  float* prelin  = (float*)alloc((size_t)BT * 256 * 4);
  float* pg1     = (float*)alloc((size_t)BT * 160 * 4);
  h16*   yin     = (h16*)  alloc((size_t)BT * 256 * 2);
  h16*   zbuf    = (h16*)  alloc((size_t)BT * 160 * 2);
  float* ybuf    = (float*)alloc((size_t)BT * 256 * 4);
  float* pg2lin  = (float*)alloc((size_t)BT * 256 * 4);
  float* obuf    = (float*)alloc((size_t)BT * 256 * 4);

  float* out_stream = (float*)d_out;
  float* out_S      = out_stream + (size_t)BB * NM * TT * DD;

  // prep
  fold_mhc_kernel<<<64, 256, 0, stream>>>(q_norm_w, q_ppre, q_ppost, q_pres, WfoldQ);
  fold_mhc_kernel<<<64, 256, 0, stream>>>(kv_norm_w, kv_ppre, kv_ppost, kv_pres, WfoldK);
  wcat_q_kernel<<<512, 256, 0, stream>>>(Wq, lora_A_q, WcatQ);
  wcat_kv_kernel<<<1536, 256, 0, stream>>>(Wk, Wv, lora_A_k, lora_A_v, alpha_up, beta_up, WcatK);
  wtail_kernel<<<256, 256, 0, stream>>>(W_pre, W_o, W_pg1, W_pg2, Bpre, Bo, Bpg1, Bpg2);
  // rms + gates
  rms_route_kernel<<<BT, 256, 0, stream>>>(st, rqw, rkw, rx, gate_q, gate_kv);
  // mHC GEMMs: (8192 x 1024) @ (1024 x 384)
  gemm_wmma_f16<4><<<dim3(6, 512, 1), 32, 0, stream>>>(rx, WfoldQ, mhcQ, 1024, 1024, 1024, 384, 0, 0, 0);
  gemm_wmma_f16<4><<<dim3(6, 512, 1), 32, 0, stream>>>(rx, WfoldK, mhcK, 1024, 1024, 1024, 384, 0, 0, 0);
  mhc_post_kernel<<<512, 256, 0, stream>>>(mhcQ, q_bpre, q_bpost, q_bres, q_apre, q_apost, q_ares);
  mhc_post_kernel<<<512, 256, 0, stream>>>(mhcK, kv_bpre, kv_bpost, kv_bres, kv_apre, kv_apost, kv_ares);
  compute_h_kernel<<<BT, 256, 0, stream>>>(st, mhcQ, mhcK, gate_q, gate_kv, h_q, h_kv,
                                           hq_r, hkv_r, Hpost_r, Hres_r);
  // per-expert projections (batched over z = expert)
  gemm_wmma_f16<2><<<dim3(1, 512, 16), 32, 0, stream>>>(h_q, WcatQ, proj_q, 256, 256, 256, 32,
                                                        (long)BT * DD, (long)32 * DD, (long)BT * 32);
  gemm_wmma_f16<2><<<dim3(3, 512, 16), 32, 0, stream>>>(h_kv, WcatK, proj_kv, 256, 256, 256, 96,
                                                        (long)BT * DD, (long)96 * DD, (long)BT * 96);
  epi_q_kernel<<<128, 64, 0, stream>>>(proj_q, gate_q, lora_B_q, q_tot);
  epi_kv_kernel<<<128, 64, 0, stream>>>(proj_kv, gate_kv, lora_B_k, lora_B_v, alpha_dn, beta_dn,
                                        pope, k_tot, v_tot, a_tot, b_tot);
  // sequential scan
  kda_kernel<<<BB, 32, 0, stream>>>(q_tot, k_tot, v_tot, a_tot, b_tot, out_kda, out_S);
  // tail GEMMs + gating
  gemm_wmma_f16<4><<<dim3(4, 512, 1), 32, 0, stream>>>(hkv_r, Bpre, prelin, 256, 256, 256, 256, 0, 0, 0);
  gemm_wmma_f16<2><<<dim3(5, 512, 1), 32, 0, stream>>>(hq_r, Bpg1, pg1, 256, 256, 256, 160, 0, 0, 0);
  yin_kernel<<<BT, 256, 0, stream>>>(prelin, out_kda, yin);
  zbuf_kernel<<<5120, 256, 0, stream>>>(pg1, zbuf);
  gemm_wmma_f16<4><<<dim3(4, 512, 1), 32, 0, stream>>>(yin, Bo, ybuf, 256, 256, 256, 256, 0, 0, 0);
  gemm_wmma_f16<4><<<dim3(4, 512, 1), 32, 0, stream>>>(zbuf, Bpg2, pg2lin, 160, 160, 160, 256, 0, 0, 0);
  ogate_kernel<<<BT, 256, 0, stream>>>(ybuf, pg2lin, obuf);
  // final write-back (stream_update) + S_new already stored by kda_kernel
  writeback_kernel<<<BT, 256, 0, stream>>>(st, obuf, Hres_r, Hpost_r, out_stream);
}
